// Attention_38560216383833
// MI455X (gfx1250) — matrix operand order repaired and checked by reading, not compiled, not run on hardware
//
#include <hip/hip_runtime.h>

// ---------------------------------------------------------------------------
// Multi-head attention forward for MI455X (gfx1250, wave32, WMMA bf16).
// B=16, N=1024, C=768, H=12, HD=64.
// ---------------------------------------------------------------------------

typedef __attribute__((ext_vector_type(16))) __bf16 bf16x16;
typedef __attribute__((ext_vector_type(8)))  __bf16 bf16x8;
typedef __attribute__((ext_vector_type(8)))  float  f32x8;

#define BB 16
#define NN 1024
#define CC 768
#define HH 12
#define HD 64
#define SCALE_F 0.125f

#define M_TOT (BB * NN)      // 16384
#define QKV_N (3 * CC)       // 2304

__device__ __forceinline__ bf16x16 join16(bf16x8 lo, bf16x8 hi) {
  return __builtin_shufflevector(lo, hi, 0, 1, 2, 3, 4, 5, 6, 7,
                                         8, 9, 10, 11, 12, 13, 14, 15);
}

__device__ __forceinline__ f32x8 wmma_bf16(bf16x16 a, bf16x16 b, f32x8 c) {
  return __builtin_amdgcn_wmma_f32_16x16x32_bf16(false, a, false, b,
                                                 (short)0, c, false, false);
}

// ---------------------------------------------------------------------------
// GEMM core: block tile 256(M) x 128(N) x 32(K), 8 waves in 4x2 grid,
// each wave owns a 64x64 output tile = 4x4 WMMA tiles (16 v_wmma / k-step,
// fed by 16 ds_load_b128 -> 1:1 LDS:WMMA ratio).
// ---------------------------------------------------------------------------

// Kernel 1: qkv = x @ W_qkv   (f32 inputs -> bf16 output [M_TOT][2304])
__global__ __launch_bounds__(256) void qkv_gemm(const float* __restrict__ X,
                                                const float* __restrict__ W,
                                                __bf16* __restrict__ Out) {
  const int K = CC, Nout = QKV_N;
  __shared__ __bf16 As[256][40];   // [m][k], padded row (20 KB)
  __shared__ __bf16 Bs[128][40];   // [n][k] transposed W tile (10 KB)

  const int tid  = threadIdx.x;
  const int lane = tid & 31, w = tid >> 5;
  const int wr = w & 3, wc = w >> 2;           // wave grid 4x2
  const int m0 = blockIdx.y * 256, n0 = blockIdx.x * 128;
  const int half = lane >> 4, li = lane & 15;

  f32x8 acc[4][4] = {};

  const int bk = tid & 31, bng = tid >> 5;     // B staging: 32 k x 8 n-groups

  for (int k0 = 0; k0 < K; k0 += 32) {
    // prefetch next k-tile while this one computes
    if (k0 + 32 < K) {
      __builtin_prefetch(X + (size_t)(m0 + tid) * K + k0 + 32, 0, 0);
      __builtin_prefetch(W + (size_t)(k0 + 32 + bk) * Nout + n0 + bng * 16, 0, 0);
    }
    __syncthreads();
    // stage A tile: one full 32-f32 row per thread -> 32 bf16 contiguous
    {
      const float4* ap = (const float4*)(X + (size_t)(m0 + tid) * K + k0);
#pragma unroll
      for (int v = 0; v < 4; ++v) {
        float4 q0 = ap[2 * v], q1 = ap[2 * v + 1];
        bf16x8 t;
        t[0] = (__bf16)q0.x; t[1] = (__bf16)q0.y;
        t[2] = (__bf16)q0.z; t[3] = (__bf16)q0.w;
        t[4] = (__bf16)q1.x; t[5] = (__bf16)q1.y;
        t[6] = (__bf16)q1.z; t[7] = (__bf16)q1.w;
        *(bf16x8*)&As[tid][v * 8] = t;
      }
    }
    // stage B tile transposed (W row-major -> Bs[n][k])
    {
      const float4* bp =
          (const float4*)(W + (size_t)(k0 + bk) * Nout + n0 + bng * 16);
#pragma unroll
      for (int v = 0; v < 4; ++v) {
        float4 q = bp[v];
        Bs[bng * 16 + v * 4 + 0][bk] = (__bf16)q.x;
        Bs[bng * 16 + v * 4 + 1][bk] = (__bf16)q.y;
        Bs[bng * 16 + v * 4 + 2][bk] = (__bf16)q.z;
        Bs[bng * 16 + v * 4 + 3][bk] = (__bf16)q.w;
      }
    }
    __syncthreads();

    bf16x16 af[4];
#pragma unroll
    for (int mi = 0; mi < 4; ++mi) {
      const __bf16* r = &As[wr * 64 + mi * 16 + li][half * 8];
      af[mi] = join16(*(const bf16x8*)r, *(const bf16x8*)(r + 16));
    }
#pragma unroll
    for (int ni = 0; ni < 4; ++ni) {
      const __bf16* r = &Bs[wc * 64 + ni * 16 + li][half * 8];
      bf16x16 bfm = join16(*(const bf16x8*)r, *(const bf16x8*)(r + 16));
#pragma unroll
      for (int mi = 0; mi < 4; ++mi)
        acc[mi][ni] = wmma_bf16(af[mi], bfm, acc[mi][ni]);
    }
  }

#pragma unroll
  for (int mi = 0; mi < 4; ++mi)
#pragma unroll
    for (int ni = 0; ni < 4; ++ni) {
      const int n  = n0 + wc * 64 + ni * 16 + li;
      const int mb = m0 + wr * 64 + mi * 16 + half * 8;
#pragma unroll
      for (int r = 0; r < 8; ++r)
        Out[(size_t)(mb + r) * Nout + n] = (__bf16)acc[mi][ni][r];
    }
}

// ---------------------------------------------------------------------------
// Kernel 2: flash attention. One block per (b, h, 128-query block).
// Computes S^T = K*Q^T (each lane owns one query column -> cheap online
// softmax: 1 shfl_xor instead of a lane tree), then O^T = V^T * P.
// ---------------------------------------------------------------------------
__global__ __launch_bounds__(256) void attn_kernel(const __bf16* __restrict__ qkv,
                                                   __bf16* __restrict__ O) {
  __shared__ __bf16 Kl[32][72];        // K tile [j][d], padded
  __shared__ __bf16 Vt[64][40];        // V tile transposed [d][j], padded
  __shared__ __bf16 Pl[8][32][16];     // per-wave P patch [j][i]

  const int tid  = threadIdx.x;
  const int lane = tid & 31, w = tid >> 5;
  const int half = lane >> 4, li = lane & 15;

  const int qb = blockIdx.x & 7;
  const int bh = blockIdx.x >> 3;
  const int b  = bh / HH, h = bh % HH;

  const size_t base = (size_t)b * NN * QKV_N;
  const int nq = qb * 128 + w * 16 + li;       // this lane's query index

  // Q held as two B-fragments (d = 0..31 and 32..63)
  const __bf16* qrow = qkv + base + (size_t)nq * QKV_N + h * HD;
  bf16x16 qf[2];
#pragma unroll
  for (int db = 0; db < 2; ++db) {
    const __bf16* p = qrow + db * 32 + half * 8;
    qf[db] = join16(*(const bf16x8*)p, *(const bf16x8*)(p + 16));
  }

  float m_run = -__builtin_inff();
  float l_run = 0.0f;
  f32x8 oacc[4] = {};

  const int sj = tid >> 3, sseg = tid & 7;     // staging: 32 rows x 8 segs

  for (int jb = 0; jb < NN; jb += 32) {
    __syncthreads();
    // stage K tile row-major
    {
      const __bf16* kr = qkv + base + (size_t)(jb + sj) * QKV_N + CC +
                         h * HD + sseg * 8;
      *(bf16x8*)&Kl[sj][sseg * 8] = *(const bf16x8*)kr;
    }
    // stage V tile transposed
    {
      const __bf16* vr = qkv + base + (size_t)(jb + sj) * QKV_N + 2 * CC +
                         h * HD + sseg * 8;
      bf16x8 vv = *(const bf16x8*)vr;
#pragma unroll
      for (int e = 0; e < 8; ++e) Vt[sseg * 8 + e][sj] = vv[e];
    }
    __syncthreads();

    // S^T tiles: D[j][i], two 16-j tiles covering 32 keys
    f32x8 st[2];
#pragma unroll
    for (int js = 0; js < 2; ++js) {
      const __bf16* kr = &Kl[js * 16 + li][half * 8];
      bf16x16 ka0 = join16(*(const bf16x8*)kr, *(const bf16x8*)(kr + 16));
      bf16x16 ka1 = join16(*(const bf16x8*)(kr + 32), *(const bf16x8*)(kr + 48));
      f32x8 z = {};
      f32x8 s = wmma_bf16(ka0, qf[0], z);
      s = wmma_bf16(ka1, qf[1], s);
      st[js] = s;
    }

    // online softmax over this lane's 16 key values (column = query li)
    float tmax = -__builtin_inff();
#pragma unroll
    for (int js = 0; js < 2; ++js)
#pragma unroll
      for (int r = 0; r < 8; ++r) {
        st[js][r] *= SCALE_F;
        tmax = fmaxf(tmax, st[js][r]);
      }
    tmax = fmaxf(tmax, __shfl_xor(tmax, 16, 32));   // combine lane halves
    const float mnew  = fmaxf(m_run, tmax);
    const float alpha = __expf(m_run - mnew);

    float lsum = 0.0f;
#pragma unroll
    for (int js = 0; js < 2; ++js)
#pragma unroll
      for (int r = 0; r < 8; ++r) {
        float p = __expf(st[js][r] - mnew);
        lsum += p;
        Pl[w][js * 16 + half * 8 + r][li] = (__bf16)p;
      }
    lsum += __shfl_xor(lsum, 16, 32);
    l_run = l_run * alpha + lsum;
    m_run = mnew;

#pragma unroll
    for (int dt = 0; dt < 4; ++dt)
#pragma unroll
      for (int r = 0; r < 8; ++r) oacc[dt][r] *= alpha;

    // re-layout P (D-frag) -> B-frag through per-wave LDS patch
    bf16x16 pb;
#pragma unroll
    for (int t = 0; t < 16; ++t) pb[t] = Pl[w][half * 16 + t][li];

    // O^T[d][i] += V^T[d][j] * P[j][i]
#pragma unroll
    for (int dt = 0; dt < 4; ++dt) {
      const __bf16* vr = &Vt[dt * 16 + li][half * 16];
      bf16x16 va = join16(*(const bf16x8*)vr, *(const bf16x8*)(vr + 8));
      oacc[dt] = wmma_bf16(va, pb, oacc[dt]);
    }
  }

  const float inv = 1.0f / l_run;
#pragma unroll
  for (int dt = 0; dt < 4; ++dt) {
    bf16x8 ov;
#pragma unroll
    for (int r = 0; r < 8; ++r) ov[r] = (__bf16)(oacc[dt][r] * inv);
    __bf16* op = O + (size_t)(b * NN + nq) * CC + h * HD + dt * 16 + half * 8;
    *(bf16x8*)op = ov;
  }
}

// ---------------------------------------------------------------------------
// Kernel 3: out = attn @ W_proj + b_proj   (bf16 A, f32 W/bias, f32 out)
// Same 256x128x32 tiling, 64x64 wave tiles.
// ---------------------------------------------------------------------------
__global__ __launch_bounds__(256) void proj_gemm(const __bf16* __restrict__ Ain,
                                                 const float* __restrict__ W,
                                                 const float* __restrict__ bias,
                                                 float* __restrict__ Out) {
  const int K = CC, Nout = CC;
  __shared__ __bf16 As[256][40];
  __shared__ __bf16 Bs[128][40];

  const int tid  = threadIdx.x;
  const int lane = tid & 31, w = tid >> 5;
  const int wr = w & 3, wc = w >> 2;
  const int m0 = blockIdx.y * 256, n0 = blockIdx.x * 128;
  const int half = lane >> 4, li = lane & 15;

  f32x8 acc[4][4] = {};

  const int bk = tid & 31, bng = tid >> 5;

  for (int k0 = 0; k0 < K; k0 += 32) {
    if (k0 + 32 < K) {
      __builtin_prefetch(Ain + (size_t)(m0 + tid) * K + k0 + 32, 0, 0);
      __builtin_prefetch(W + (size_t)(k0 + 32 + bk) * Nout + n0 + bng * 16, 0, 0);
    }
    __syncthreads();
    // stage A tile: one full 32-bf16 row per thread (raw copy)
    {
      const bf16x8* ap = (const bf16x8*)(Ain + (size_t)(m0 + tid) * K + k0);
#pragma unroll
      for (int v = 0; v < 4; ++v) *(bf16x8*)&As[tid][v * 8] = ap[v];
    }
    {
      const float4* bp =
          (const float4*)(W + (size_t)(k0 + bk) * Nout + n0 + bng * 16);
#pragma unroll
      for (int v = 0; v < 4; ++v) {
        float4 q = bp[v];
        Bs[bng * 16 + v * 4 + 0][bk] = (__bf16)q.x;
        Bs[bng * 16 + v * 4 + 1][bk] = (__bf16)q.y;
        Bs[bng * 16 + v * 4 + 2][bk] = (__bf16)q.z;
        Bs[bng * 16 + v * 4 + 3][bk] = (__bf16)q.w;
      }
    }
    __syncthreads();

    bf16x16 af[4];
#pragma unroll
    for (int mi = 0; mi < 4; ++mi) {
      const __bf16* r = &As[wr * 64 + mi * 16 + li][half * 8];
      af[mi] = join16(*(const bf16x8*)r, *(const bf16x8*)(r + 16));
    }
#pragma unroll
    for (int ni = 0; ni < 4; ++ni) {
      const __bf16* r = &Bs[wc * 64 + ni * 16 + li][half * 8];
      bf16x16 bfm = join16(*(const bf16x8*)r, *(const bf16x8*)(r + 16));
#pragma unroll
      for (int mi = 0; mi < 4; ++mi)
        acc[mi][ni] = wmma_bf16(af[mi], bfm, acc[mi][ni]);
    }
  }

#pragma unroll
  for (int mi = 0; mi < 4; ++mi)
#pragma unroll
    for (int ni = 0; ni < 4; ++ni) {
      const int n  = n0 + wc * 64 + ni * 16 + li;
      const int mb = m0 + wr * 64 + mi * 16 + half * 8;
      const float bn = bias[n];
#pragma unroll
      for (int r = 0; r < 8; ++r)
        Out[(size_t)(mb + r) * Nout + n] = acc[mi][ni][r] + bn;
    }
}

// ---------------------------------------------------------------------------
extern "C" void kernel_launch(void* const* d_in, const int* in_sizes, int n_in,
                              void* d_out, int out_size, void* d_ws, size_t ws_size,
                              hipStream_t stream) {
  (void)in_sizes; (void)n_in; (void)out_size; (void)ws_size;

  const float* x      = (const float*)d_in[0];   // [16,1024,768]
  const float* W_qkv  = (const float*)d_in[1];   // [768,2304]
  const float* W_proj = (const float*)d_in[2];   // [768,768]
  const float* b_proj = (const float*)d_in[3];   // [768]
  float* out = (float*)d_out;                    // [16,1024,768]

  __bf16* qkv_bf  = (__bf16*)d_ws;                                      // 75.5 MB
  __bf16* attn_bf = (__bf16*)((char*)d_ws + (size_t)M_TOT * QKV_N * 2); // 25.2 MB

  qkv_gemm<<<dim3(QKV_N / 128, M_TOT / 256), 256, 0, stream>>>(x, W_qkv, qkv_bf);
  attn_kernel<<<dim3(BB * HH * (NN / 128)), 256, 0, stream>>>(qkv_bf, attn_bf);
  proj_gemm<<<dim3(CC / 128, M_TOT / 256), 256, 0, stream>>>(attn_bf, W_proj,
                                                             b_proj, out);
}
